// MultiheadFlashDiff1_53360673685727
// MI455X (gfx1250) — matrix operand order repaired and checked by reading, not compiled, not run on hardware
//
#include <hip/hip_runtime.h>

// ---------------------------------------------------------------------------
// MI455X (gfx1250) DiffAttn implementation: bf16 WMMA everywhere.
// ---------------------------------------------------------------------------

typedef __bf16 bf16_t;
typedef __attribute__((ext_vector_type(16))) __bf16 bf16x16;
typedef __attribute__((ext_vector_type(8)))  float  f32x8;
typedef __attribute__((ext_vector_type(4)))  unsigned int u32x4;
typedef __attribute__((ext_vector_type(4)))  int i32x4;

#define SEQ  2048
#define NB   2
#define NH   16
#define EMB  2048
#define MROWS (NB * SEQ)          // 4096
#define LAMBDA_INIT_C 0.78360577f // 0.8 - 0.6*exp(-0.3*12)

// ---- CDNA5 async global->LDS path (builtin confirmed present; typed int4
// pointers: AS1 source, AS3 LDS destination) -------------------------------
#if defined(__has_builtin)
#if __has_builtin(__builtin_amdgcn_global_load_async_to_lds_b128)
#define HAVE_ASYNC_LDS 1
#endif
#endif
#ifndef HAVE_ASYNC_LDS
#define HAVE_ASYNC_LDS 0
#endif

#if HAVE_ASYNC_LDS
#define AS1P(p) ((__attribute__((address_space(1))) i32x4*)(void*)(p))
#define AS3P(p) ((__attribute__((address_space(3))) i32x4*)(void*)(p))
__device__ __forceinline__ void wait_asynccnt0() {
#if __has_builtin(__builtin_amdgcn_s_wait_asynccnt)
  __builtin_amdgcn_s_wait_asynccnt(0);
#else
  asm volatile("s_wait_asynccnt 0x0" ::: "memory");
#endif
}
#endif

__device__ __forceinline__ f32x8 wmma_bf16f32(bf16x16 a, bf16x16 b, f32x8 c) {
  // v_wmma_f32_16x16x32_bf16 (D = A*B + C), wave32
  return __builtin_amdgcn_wmma_f32_16x16x32_bf16(false, a, false, b, (short)0, c,
                                                 false, false);
}

// ---------------------------------------------------------------------------
// fp32 -> bf16 conversion
// ---------------------------------------------------------------------------
__global__ void cvt_f32_bf16(const float* __restrict__ in, bf16_t* __restrict__ out,
                             size_t n) {
  size_t i = (size_t)blockIdx.x * blockDim.x + threadIdx.x;
  if (i < n) out[i] = (bf16_t)in[i];
}

// ---------------------------------------------------------------------------
// bf16 WMMA GEMM: C(MxN) = A(MxK) * B(KxN).  BM=BN=128, BK=32, 8 waves,
// double-buffered LDS with one barrier per K-step; next panel's global
// traffic (async DMA for A, register-staged transpose for B) is issued
// before the current panel's WMMAs so HBM latency hides behind matrix math.
// blockIdx.z selects one of three (B, C) pairs (fused QKV).
// ---------------------------------------------------------------------------
template <typename OutT>
__global__ __launch_bounds__(256) void gemm_wmma(
    const bf16_t* __restrict__ A,
    const bf16_t* __restrict__ B0, const bf16_t* __restrict__ B1,
    const bf16_t* __restrict__ B2,
    OutT* __restrict__ C0, OutT* __restrict__ C1, OutT* __restrict__ C2,
    int M, int N, int K) {
  const bf16_t* Bm = (blockIdx.z == 0) ? B0 : (blockIdx.z == 1) ? B1 : B2;
  OutT* Cm         = (blockIdx.z == 0) ? C0 : (blockIdx.z == 1) ? C1 : C2;

  __shared__ __align__(16) bf16_t As[2][128][32];  // [buf][m][k]
  __shared__ __align__(16) bf16_t Bs[2][128][32];  // [buf][n][k] (K-transposed)

  const int tid  = threadIdx.x;
  const int wave = tid >> 5;
  const int lane = tid & 31;
  const int hl   = lane >> 4;   // lane half (K-select for A/B frags)
  const int l16  = lane & 15;

  const int wm = (wave >> 1) * 32;  // 4 wave rows
  const int wn = (wave & 1) * 64;   // 2 wave cols
  const int m0 = blockIdx.x * 128;
  const int n0 = blockIdx.y * 128;

  // per-thread load assignments
  const int ar = tid >> 1;          // A: 2 threads/row, 16 bf16 each
  const int ac = (tid & 1) << 4;
  const int bk = tid >> 3;          // B: 8 threads/row(k), 16 bf16 along n
  const int bn = (tid & 7) << 4;

  f32x8 acc[2][4];
#pragma unroll
  for (int i = 0; i < 2; ++i)
#pragma unroll
    for (int j = 0; j < 4; ++j)
#pragma unroll
      for (int r = 0; r < 8; ++r) acc[i][j][r] = 0.0f;

  u32x4 areg[2], breg[2];

  auto issue_a = [&](int k0, int buf) {
    const bf16_t* src = A + (size_t)(m0 + ar) * K + k0 + ac;
#if HAVE_ASYNC_LDS
    // DMA straight into LDS, tracked by ASYNCcnt (no VGPR round-trip)
    __builtin_amdgcn_global_load_async_to_lds_b128(AS1P(src),
                                                   AS3P(&As[buf][ar][ac]), 0, 0);
    __builtin_amdgcn_global_load_async_to_lds_b128(
        AS1P(src + 8), AS3P(&As[buf][ar][ac + 8]), 0, 0);
#else
    const u32x4* s = reinterpret_cast<const u32x4*>(src);
    areg[0] = s[0];
    areg[1] = s[1];
#endif
  };
  auto issue_b = [&](int k0) {
    const u32x4* s =
        reinterpret_cast<const u32x4*>(Bm + (size_t)(k0 + bk) * N + n0 + bn);
    breg[0] = s[0];
    breg[1] = s[1];
  };
  auto commit = [&](int buf) {
#if !HAVE_ASYNC_LDS
    u32x4* ad = reinterpret_cast<u32x4*>(&As[buf][ar][ac]);
    ad[0] = areg[0];
    ad[1] = areg[1];
#endif
    union { u32x4 q[2]; bf16_t h[16]; } u;
    u.q[0] = breg[0];
    u.q[1] = breg[1];
#pragma unroll
    for (int j = 0; j < 16; ++j) Bs[buf][bn + j][bk] = u.h[j];
#if HAVE_ASYNC_LDS
    wait_asynccnt0();
#endif
  };

  // prologue: panel 0
  issue_a(0, 0);
  issue_b(0);
  commit(0);
  __syncthreads();

  const int T = K >> 5;  // K/32 panels
  for (int t = 0; t < T; ++t) {
    const int cur = t & 1;
    const int nxt = cur ^ 1;
    if (t + 1 < T) {  // start next panel's global traffic first
      issue_a((t + 1) << 5, nxt);
      issue_b((t + 1) << 5);
    }

    // --- A fragments: lane=row, elems j -> K=hl*16+j (contiguous)
    bf16x16 afr[2];
#pragma unroll
    for (int i = 0; i < 2; ++i) {
      const bf16_t* p = &As[cur][wm + i * 16 + l16][hl << 4];
      bf16x16 f;
#pragma unroll
      for (int j = 0; j < 8; ++j) { f[j] = p[j]; f[j + 8] = p[j + 8]; }
      afr[i] = f;
    }
    // --- B fragments: lane=col, elems j -> K=hl*16+j (contiguous in Bs) ---
    bf16x16 bfr[4];
#pragma unroll
    for (int j = 0; j < 4; ++j) {
      const bf16_t* p = &Bs[cur][wn + j * 16 + l16][hl << 4];
      bf16x16 f;
#pragma unroll
      for (int e = 0; e < 16; ++e) f[e] = p[e];
      bfr[j] = f;
    }
#pragma unroll
    for (int i = 0; i < 2; ++i)
#pragma unroll
      for (int j = 0; j < 4; ++j)
        acc[i][j] = wmma_bf16f32(afr[i], bfr[j], acc[i][j]);

    if (t + 1 < T) commit(nxt);  // drain loads into the other buffer
    __syncthreads();             // single barrier per K-step
  }

  // --- store: C/D layout: row = base + r + 8*hl, col = base + l16 ---
#pragma unroll
  for (int i = 0; i < 2; ++i) {
    const int rbase = m0 + wm + i * 16 + (hl << 3);
#pragma unroll
    for (int j = 0; j < 4; ++j) {
      const int col = n0 + wn + j * 16 + l16;
#pragma unroll
      for (int r = 0; r < 8; ++r)
        Cm[(size_t)(rbase + r) * N + col] = (OutT)acc[i][j][r];
    }
  }
}

// ---------------------------------------------------------------------------
// Interleaved RoPE in-place on q (y==0) / k (y==1): (B,S,32,64), pairs of 2.
// ---------------------------------------------------------------------------
__global__ void rope_kernel(bf16_t* __restrict__ q, bf16_t* __restrict__ k) {
  const size_t idx = (size_t)blockIdx.x * blockDim.x + threadIdx.x;
  const size_t npairs = (size_t)NB * SEQ * 32 * 32;
  if (idx >= npairs) return;
  bf16_t* buf = (blockIdx.y == 0) ? q : k;
  const int pair   = (int)(idx & 31);
  const size_t row = idx >> 5;                 // (b*SEQ + t)*32 + head
  const int t      = (int)((row >> 5) % SEQ);
  const float invf = __powf(10000.0f, -(float)pair * (1.0f / 32.0f));
  float sn, cs;
  __sincosf((float)t * invf, &sn, &cs);
  bf16_t* p = buf + row * 64 + pair * 2;
  const float x1 = (float)p[0];
  const float x2 = (float)p[1];
  p[0] = (bf16_t)(x1 * cs - x2 * sn);
  p[1] = (bf16_t)(x1 * sn + x2 * cs);
}

// ---------------------------------------------------------------------------
// Online-softmax update for one 16x64 score block (4 n-tiles of 16x16 f32).
// Row stats per lane: row = r + 8*hl; reduce across the 16 lanes of a half.
// ---------------------------------------------------------------------------
__device__ __forceinline__ void softmax_block(f32x8* s, float* rm, float* rl,
                                              f32x8* o, bool diag, int kbase,
                                              int rowbase, int hl, int l16) {
  const float sc = 0.125f;  // 64^-0.5
#pragma unroll
  for (int nt = 0; nt < 4; ++nt) {
    const int key = kbase + nt * 16 + l16;
#pragma unroll
    for (int r = 0; r < 8; ++r) {
      float v = s[nt][r] * sc;
      if (diag && key > rowbase + r + (hl << 3)) v = -1e30f;  // causal mask
      s[nt][r] = v;
    }
  }
#pragma unroll
  for (int r = 0; r < 8; ++r) {
    float m = fmaxf(fmaxf(s[0][r], s[1][r]), fmaxf(s[2][r], s[3][r]));
#pragma unroll
    for (int off = 1; off < 16; off <<= 1) m = fmaxf(m, __shfl_xor(m, off, 32));
    const float mnew = fmaxf(rm[r], m);
    const float resc = __expf(rm[r] - mnew);
    rm[r] = mnew;
    rl[r] *= resc;
#pragma unroll
    for (int t = 0; t < 8; ++t) o[t][r] *= resc;
    float su = 0.0f;
#pragma unroll
    for (int nt = 0; nt < 4; ++nt) {
      const float p = __expf(s[nt][r] - mnew);
      s[nt][r] = p;
      su += p;
    }
#pragma unroll
    for (int off = 1; off < 16; off <<= 1) su += __shfl_xor(su, off, 32);
    rl[r] += su;
  }
}

// ---------------------------------------------------------------------------
// Differential flash attention.  grid=(32 qblocks, 16 heads, 2 batch),
// block=128 (4 waves); each wave owns 16 query rows for BOTH attn1/attn2.
// q,k: (B,S,32,64) bf16 (heads 2h / 2h+1);  v: (B,S,16,128) bf16.
// Output: (B,S,16,128) bf16 = diff + RMSnorm + ln_w + (1-lambda_init).
// ---------------------------------------------------------------------------
__global__ __launch_bounds__(128) void diff_attn_kernel(
    const bf16_t* __restrict__ q, const bf16_t* __restrict__ kt,
    const bf16_t* __restrict__ vt, const float* __restrict__ lq1,
    const float* __restrict__ lk1, const float* __restrict__ lq2,
    const float* __restrict__ lk2, const float* __restrict__ lnw,
    bf16_t* __restrict__ attn_out) {
  const int qb = blockIdx.x;
  const int h  = blockIdx.y;
  const int b  = blockIdx.z;
  const int tid  = threadIdx.x;
  const int wave = tid >> 5;
  const int lane = tid & 31;
  const int hl   = lane >> 4;
  const int l16  = lane & 15;

  __shared__ __align__(16) bf16_t K1s[64][64];        // 8 KB
  __shared__ __align__(16) bf16_t K2s[64][64];        // 8 KB
  __shared__ __align__(16) bf16_t Vts[128][64];       // 16 KB (V transposed)
  __shared__ __align__(16) bf16_t Ps[4][2][16][64];   // 16 KB  P staging

  // lambda_full (tiny dot products, uniform across lanes)
  float d1 = 0.0f, d2 = 0.0f;
  for (int i = 0; i < 64; ++i) {
    d1 += lq1[i] * lk1[i];
    d2 += lq2[i] * lk2[i];
  }
  const float lambda_full = __expf(d1) - __expf(d2) + LAMBDA_INIT_C;

  const int rowbase = qb * 64 + wave * 16;
  const int mrow    = rowbase + l16;
  const bf16_t* q1p = q + (((size_t)b * SEQ + mrow) * 32 + 2 * h) * 64;
  const bf16_t* q2p = q1p + 64;

  // Q fragments (A layout), two K-chunks of 32 each
  bf16x16 a1[2], a2[2];
#pragma unroll
  for (int c = 0; c < 2; ++c) {
    const bf16_t* p1 = q1p + c * 32 + (hl << 4);
    const bf16_t* p2 = q2p + c * 32 + (hl << 4);
    bf16x16 f1, f2;
#pragma unroll
    for (int j = 0; j < 8; ++j) {
      f1[j] = p1[j]; f1[j + 8] = p1[j + 8];
      f2[j] = p2[j]; f2[j + 8] = p2[j + 8];
    }
    a1[c] = f1;
    a2[c] = f2;
  }

  f32x8 o1[8], o2[8];
  float rm1[8], rm2[8], rl1[8], rl2[8];
#pragma unroll
  for (int t = 0; t < 8; ++t)
#pragma unroll
    for (int r = 0; r < 8; ++r) { o1[t][r] = 0.0f; o2[t][r] = 0.0f; }
#pragma unroll
  for (int r = 0; r < 8; ++r) { rm1[r] = -1e30f; rm2[r] = -1e30f; rl1[r] = 0.0f; rl2[r] = 0.0f; }

  for (int kb = 0; kb <= qb; ++kb) {
    const int kbase = kb * 64;
    // --- K1/K2 tiles 64x64: 2 threads/row, 32 bf16 (64B) each ---
    {
      const int r  = tid >> 1;
      const int c0 = (tid & 1) << 5;
      const bf16_t* s1 =
          kt + (((size_t)b * SEQ + kbase + r) * 32 + 2 * h) * 64 + c0;
#if HAVE_ASYNC_LDS
#pragma unroll
      for (int j = 0; j < 4; ++j) {
        __builtin_amdgcn_global_load_async_to_lds_b128(
            AS1P(s1 + j * 8), AS3P(&K1s[r][c0 + j * 8]), 0, 0);
        __builtin_amdgcn_global_load_async_to_lds_b128(
            AS1P(s1 + 64 + j * 8), AS3P(&K2s[r][c0 + j * 8]), 0, 0);
      }
#else
      const u32x4* u1 = reinterpret_cast<const u32x4*>(s1);
      const u32x4* u2 = reinterpret_cast<const u32x4*>(s1 + 64);
      u32x4* dK1 = reinterpret_cast<u32x4*>(&K1s[r][c0]);
      u32x4* dK2 = reinterpret_cast<u32x4*>(&K2s[r][c0]);
#pragma unroll
      for (int j = 0; j < 4; ++j) { dK1[j] = u1[j]; dK2[j] = u2[j]; }
#endif
    }
    // --- V tile 64x128 -> transposed Vts[col][key] ---
    {
      const int key = tid & 63;
      const int ch  = tid >> 6;
      const bf16_t* vs =
          vt + (((size_t)b * SEQ + kbase + key) * NH + h) * 128 + (ch << 6);
      union { u32x4 u[8]; bf16_t hh[64]; } vb;
      const u32x4* uv = reinterpret_cast<const u32x4*>(vs);
#pragma unroll
      for (int j = 0; j < 8; ++j) vb.u[j] = uv[j];
#pragma unroll
      for (int j = 0; j < 64; ++j) Vts[(ch << 6) + j][key] = vb.hh[j];
    }
    // prefetch next kv block (global_prefetch_b8)
    if (kb + 1 <= qb) {
      __builtin_prefetch(
          kt + (((size_t)b * SEQ + kbase + 64 + (tid >> 1)) * 32 + 2 * h) * 64,
          0, 1);
      __builtin_prefetch(
          vt + (((size_t)b * SEQ + kbase + 64 + (tid & 63)) * NH + h) * 128, 0,
          1);
    }
#if HAVE_ASYNC_LDS
    wait_asynccnt0();  // drain async K-tile DMA before the barrier
#endif
    __syncthreads();

    // --- scores: S = Q * K^T (B frag reads K rows contiguously) ---
    f32x8 s1[4], s2[4];
#pragma unroll
    for (int nt = 0; nt < 4; ++nt)
#pragma unroll
      for (int r = 0; r < 8; ++r) { s1[nt][r] = 0.0f; s2[nt][r] = 0.0f; }
#pragma unroll
    for (int nt = 0; nt < 4; ++nt) {
      const int keyl = nt * 16 + l16;
#pragma unroll
      for (int c = 0; c < 2; ++c) {
        const bf16_t* kp1 = &K1s[keyl][c * 32 + (hl << 4)];
        const bf16_t* kp2 = &K2s[keyl][c * 32 + (hl << 4)];
        bf16x16 f1, f2;
#pragma unroll
        for (int j = 0; j < 16; ++j) { f1[j] = kp1[j]; f2[j] = kp2[j]; }
        s1[nt] = wmma_bf16f32(a1[c], f1, s1[nt]);
        s2[nt] = wmma_bf16f32(a2[c], f2, s2[nt]);
      }
    }

    const bool diag = (kb == qb);
    softmax_block(s1, rm1, rl1, o1, diag, kbase, rowbase, hl, l16);
    softmax_block(s2, rm2, rl2, o2, diag, kbase, rowbase, hl, l16);

    // --- stage P (C layout -> LDS -> A layout), per-wave private region ---
#pragma unroll
    for (int nt = 0; nt < 4; ++nt)
#pragma unroll
      for (int r = 0; r < 8; ++r) {
        Ps[wave][0][r + (hl << 3)][nt * 16 + l16] = (bf16_t)s1[nt][r];
        Ps[wave][1][r + (hl << 3)][nt * 16 + l16] = (bf16_t)s2[nt][r];
      }
    bf16x16 p1f[2], p2f[2];
#pragma unroll
    for (int c = 0; c < 2; ++c) {
      const bf16_t* pp1 = &Ps[wave][0][l16][c * 32 + (hl << 4)];
      const bf16_t* pp2 = &Ps[wave][1][l16][c * 32 + (hl << 4)];
      bf16x16 f1, f2;
#pragma unroll
      for (int j = 0; j < 8; ++j) {
        f1[j] = pp1[j]; f1[j + 8] = pp1[j + 8];
        f2[j] = pp2[j]; f2[j + 8] = pp2[j + 8];
      }
      p1f[c] = f1;
      p2f[c] = f2;
    }
    // --- O += P * V  (V fragments shared by attn1/attn2) ---
#pragma unroll
    for (int nt = 0; nt < 8; ++nt) {
#pragma unroll
      for (int c = 0; c < 2; ++c) {
        const bf16_t* vp = &Vts[nt * 16 + l16][c * 32 + (hl << 4)];
        bf16x16 vf;
#pragma unroll
        for (int j = 0; j < 16; ++j) vf[j] = vp[j];
        o1[nt] = wmma_bf16f32(p1f[c], vf, o1[nt]);
        o2[nt] = wmma_bf16f32(p2f[c], vf, o2[nt]);
      }
    }
    __syncthreads();
  }

  // --- finalize: diff, RMS-norm over 128, ln_w, (1-lambda_init) ---
  float res[8][8];
  float rmsv[8];
#pragma unroll
  for (int r = 0; r < 8; ++r) {
    const float inv1 = 1.0f / rl1[r];
    const float inv2 = 1.0f / rl2[r];
    float ss = 0.0f;
#pragma unroll
    for (int nt = 0; nt < 8; ++nt) {
      const float x = o1[nt][r] * inv1 - lambda_full * (o2[nt][r] * inv2);
      res[nt][r] = x;
      ss += x * x;
    }
#pragma unroll
    for (int off = 1; off < 16; off <<= 1) ss += __shfl_xor(ss, off, 32);
    rmsv[r] = rsqrtf(ss * (1.0f / 128.0f) + 1e-5f);
  }
  const float osc = 1.0f - LAMBDA_INIT_C;
#pragma unroll
  for (int nt = 0; nt < 8; ++nt) {
    const int col = nt * 16 + l16;
    const float w = lnw[col] * osc;
#pragma unroll
    for (int r = 0; r < 8; ++r) {
      const int trow = rowbase + r + (hl << 3);
      attn_out[(((size_t)b * SEQ + trow) * NH + h) * 128 + col] =
          (bf16_t)(res[nt][r] * rmsv[r] * w);
    }
  }
}

// ---------------------------------------------------------------------------
// Launcher
// ---------------------------------------------------------------------------
extern "C" void kernel_launch(void* const* d_in, const int* in_sizes, int n_in,
                              void* d_out, int out_size, void* d_ws,
                              size_t ws_size, hipStream_t stream) {
  const float* x   = (const float*)d_in[0];
  const float* Wq  = (const float*)d_in[1];
  const float* Wk  = (const float*)d_in[2];
  const float* Wv  = (const float*)d_in[3];
  const float* Wo  = (const float*)d_in[4];
  const float* lq1 = (const float*)d_in[5];
  const float* lk1 = (const float*)d_in[6];
  const float* lq2 = (const float*)d_in[7];
  const float* lk2 = (const float*)d_in[8];
  const float* lnw = (const float*)d_in[9];
  float* out = (float*)d_out;

  const size_t NX = (size_t)MROWS * EMB;  // 8M activations
  const size_t NW = (size_t)EMB * EMB;    // 4M weights

  char* ws = (char*)d_ws;
  bf16_t* xb    = (bf16_t*)ws;  ws += NX * sizeof(bf16_t);
  bf16_t* wqb   = (bf16_t*)ws;  ws += NW * sizeof(bf16_t);
  bf16_t* wkb   = (bf16_t*)ws;  ws += NW * sizeof(bf16_t);
  bf16_t* wvb   = (bf16_t*)ws;  ws += NW * sizeof(bf16_t);
  bf16_t* wob   = (bf16_t*)ws;  ws += NW * sizeof(bf16_t);
  bf16_t* qbuf  = (bf16_t*)ws;  ws += NX * sizeof(bf16_t);
  bf16_t* kbuf  = (bf16_t*)ws;  ws += NX * sizeof(bf16_t);
  bf16_t* vbuf  = (bf16_t*)ws;  ws += NX * sizeof(bf16_t);
  bf16_t* attnb = (bf16_t*)ws;  ws += NX * sizeof(bf16_t);

  // 1) fp32 -> bf16
  cvt_f32_bf16<<<(NX + 255) / 256, 256, 0, stream>>>(x, xb, NX);
  cvt_f32_bf16<<<(NW + 255) / 256, 256, 0, stream>>>(Wq, wqb, NW);
  cvt_f32_bf16<<<(NW + 255) / 256, 256, 0, stream>>>(Wk, wkb, NW);
  cvt_f32_bf16<<<(NW + 255) / 256, 256, 0, stream>>>(Wv, wvb, NW);
  cvt_f32_bf16<<<(NW + 255) / 256, 256, 0, stream>>>(Wo, wob, NW);

  // 2) fused QKV GEMM (z picks Wq/Wk/Wv)
  gemm_wmma<bf16_t><<<dim3(MROWS / 128, EMB / 128, 3), 256, 0, stream>>>(
      xb, wqb, wkb, wvb, qbuf, kbuf, vbuf, MROWS, EMB, EMB);

  // 3) RoPE on q and k (y picks tensor)
  {
    const size_t npairs = (size_t)NB * SEQ * 32 * 32;  // 4M
    rope_kernel<<<dim3((unsigned)((npairs + 255) / 256), 2), 256, 0, stream>>>(
        qbuf, kbuf);
  }

  // 4) differential flash attention
  diff_attn_kernel<<<dim3(SEQ / 64, NH, NB), 128, 0, stream>>>(
      qbuf, kbuf, vbuf, lq1, lk1, lq2, lk2, lnw, attnb);

  // 5) output GEMM (fp32 out)
  gemm_wmma<float><<<dim3(MROWS / 128, EMB / 128, 1), 256, 0, stream>>>(
      attnb, wob, wob, wob, out, out, out, MROWS, EMB, EMB);
}
